// DCGRUCell_3676492005530
// MI455X (gfx1250) — matrix operand order repaired and checked by reading, not compiled, not run on hardware
//
#include <hip/hip_runtime.h>
#include <hip/hip_bf16.h>

// ---------------- problem constants (from reference) ----------------
static const int kB    = 32;     // batch
static const int kN    = 1024;   // nodes
static const int kCin  = 2;
static const int kH    = 32;
static const int kS    = 2;      // supports
static const int kK    = 2;      // diffusion steps
static const int CCAT  = kCin + kH;        // 34
static const int CPAD  = 48;               // 3 x 16 col tiles
static const int NMAT  = kS * kK + 1;      // 5
static const int FTOT  = NMAT * CCAT;      // 170
static const int HPAD  = 176;              // padded h width
static const int ALD   = 36;               // LDS A row stride (f32): 16B aligned, bank-rotated

typedef __attribute__((ext_vector_type(16))) __bf16 v16bf;
typedef __attribute__((ext_vector_type(8)))  float  v8f;

// ---- CDNA5 async global->LDS path (probe-confirmed present in this toolchain) ----
#if defined(__AMDGCN__) && __has_builtin(__builtin_amdgcn_global_load_async_to_lds_b128) && \
    __has_builtin(__builtin_amdgcn_s_wait_asynccnt)
#define USE_ASYNC_LDS 1
#pragma message("CDNA5 async global->LDS builtins: ENABLED")
typedef __attribute__((ext_vector_type(4))) int v4i;
typedef __attribute__((address_space(1))) v4i* gv4i_p;   // global (__device__) b128 ptr
typedef __attribute__((address_space(3))) v4i* lv4i_p;   // LDS (__shared__) b128 ptr
#else
#define USE_ASYNC_LDS 0
#pragma message("CDNA5 async global->LDS builtins: not found, using sync staging")
#endif

// ---------------- concat: build x = [inputs, (r*)states] ----------------
__global__ __launch_bounds__(256) void concat_kernel(
    const float* __restrict__ inputs, const float* __restrict__ states,
    const float* __restrict__ ru,   // nullptr -> plain states, else r = ru[:, 0:32]
    float* __restrict__ h, __bf16* __restrict__ xbf)
{
    int idx = blockIdx.x * blockDim.x + threadIdx.x;   // b*N + n
    if (idx >= kB * kN) return;
    const float* in2 = inputs + (size_t)idx * kCin;
    const float* st  = states + (size_t)idx * kH;
    const float* r   = ru ? (ru + (size_t)idx * (2 * kH)) : nullptr;
    float*  hrow = h   + (size_t)idx * HPAD;
    __bf16* xrow = xbf + (size_t)idx * CPAD;
    #pragma unroll
    for (int c = 0; c < CPAD; ++c) {
        float v;
        if (c < kCin)       v = in2[c];
        else if (c < CCAT)  { float sv = st[c - kCin]; v = r ? (r[c - kCin] * sv) : sv; }
        else                v = 0.0f;
        if (c < CCAT) hrow[c] = v;
        xrow[c] = (__bf16)v;
    }
}

// ---------------- diffusion GEMM step: out = A[b,s] @ src ----------------
// one wave -> 16 rows x 48 cols, 8 waves/block -> 128 rows, grid (N/128, S, B)
__global__ __launch_bounds__(256) void gemm_step_kernel(
    const float*  __restrict__ supports,  // (B,S,N,N) f32
    const __bf16* __restrict__ src,       // (B,[S],N,CPAD) bf16
    int src_has_s,
    float*  __restrict__ h,               // (B,N,HPAD) f32
    __bf16* __restrict__ dst,             // (B,S,N,CPAD) bf16
    int kstep)
{
    // A panel: 128 rows x 32 f32, padded row stride 36 (16B aligned, rotates banks)
    __shared__ __attribute__((aligned(16))) float  ldsA[128 * ALD];   // 18 KB
    // B panel chunk transposed [c][k] so each lane's v16bf fragment is one 32B load
    __shared__ __attribute__((aligned(32))) __bf16 ldsB[CPAD * 32];   // 3 KB

    const int tid  = threadIdx.x;
    const int wave = tid >> 5;
    const int lane = tid & 31;
    const int half = lane >> 4;       // 0: lanes 0-15, 1: lanes 16-31
    const int ml   = lane & 15;

    const int mg = blockIdx.x;        // 0..7
    const int s  = blockIdx.y;        // 0..1
    const int b  = blockIdx.z;        // 0..31
    const int row_base = mg * 128 + wave * 16;

    const float*  Abase = supports + ((size_t)(b * kS + s)) * (size_t)kN * kN;
    const __bf16* srcp  = src + ((size_t)(src_has_s ? (b * kS + s) : b)) * (size_t)kN * CPAD;

    v8f acc0 = {}; v8f acc1 = {}; v8f acc2 = {};

    // B staging assignment: 256 threads cover 32 (k) x 48 (c): 8 col-groups of 6
    const int m_local = tid & 31;
    const int cg      = tid >> 5;     // 0..7

    for (int kb = 0; kb < kN; kb += 32) {
        // ---- stage A panel: 128 rows x 32 f32 = 1024 x b128 transfers (4/thread) ----
        #pragma unroll
        for (int j = 0; j < 4; ++j) {
            int id  = tid + 256 * j;
            int r   = id >> 3;        // 0..127
            int seg = id & 7;         // 16B segment within the 128B row chunk
            const float* g = Abase + (size_t)(mg * 128 + r) * kN + kb + seg * 4;
#if USE_ASYNC_LDS
            __builtin_amdgcn_global_load_async_to_lds_b128(
                (gv4i_p)(g),
                (lv4i_p)(&ldsA[r * ALD + seg * 4]),
                0, 0);
#else
            *(float4*)&ldsA[r * ALD + seg * 4] = *(const float4*)g;
#endif
        }

        // ---- stage B-panel chunk (32 x 48 bf16) transposed into LDS [c][k] ----
        {
            const __bf16* row = srcp + (size_t)(kb + m_local) * CPAD + cg * 6;
            __builtin_prefetch(row + 32 * CPAD, 0, 0);   // next chunk -> global_prefetch_b8
            #pragma unroll
            for (int i = 0; i < 6; ++i)
                ldsB[(cg * 6 + i) * 32 + m_local] = row[i];
        }
#if USE_ASYNC_LDS
        __builtin_amdgcn_s_wait_asynccnt(0);
#endif
        __syncthreads();

        // ---- A fragment out of LDS: rows wave*16+ml, K runs half*16..half*16+15 ----
        const float* ar = &ldsA[(wave * 16 + ml) * ALD + half * 16];
        float4 f0 = *(const float4*)(ar);
        float4 f1 = *(const float4*)(ar + 4);
        float4 f2 = *(const float4*)(ar + 8);
        float4 f3 = *(const float4*)(ar + 12);

        v16bf a;
        a[0]  = (__bf16)f0.x; a[1]  = (__bf16)f0.y; a[2]  = (__bf16)f0.z; a[3]  = (__bf16)f0.w;
        a[4]  = (__bf16)f1.x; a[5]  = (__bf16)f1.y; a[6]  = (__bf16)f1.z; a[7]  = (__bf16)f1.w;
        a[8]  = (__bf16)f2.x; a[9]  = (__bf16)f2.y; a[10] = (__bf16)f2.z; a[11] = (__bf16)f2.w;
        a[12] = (__bf16)f3.x; a[13] = (__bf16)f3.y; a[14] = (__bf16)f3.z; a[15] = (__bf16)f3.w;

        // ---- 3 B fragments (one per 16-col tile), contiguous 32B LDS loads ----
        v16bf b0 = *(const v16bf*)&ldsB[(0 * 16 + ml) * 32 + half * 16];
        v16bf b1 = *(const v16bf*)&ldsB[(1 * 16 + ml) * 32 + half * 16];
        v16bf b2 = *(const v16bf*)&ldsB[(2 * 16 + ml) * 32 + half * 16];

        acc0 = __builtin_amdgcn_wmma_f32_16x16x32_bf16(false, a, false, b0, (short)0, acc0, false, false);
        acc1 = __builtin_amdgcn_wmma_f32_16x16x32_bf16(false, a, false, b1, (short)0, acc1, false, false);
        acc2 = __builtin_amdgcn_wmma_f32_16x16x32_bf16(false, a, false, b2, (short)0, acc2, false, false);

        __syncthreads();
    }

    // ---- write out: f32 slice in h + bf16 panel for next step ----
    const int mat     = 1 + s * kK + kstep;
    const int colbase = mat * CCAT;
    #pragma unroll
    for (int v = 0; v < 8; ++v) {
        int n_glob = row_base + v + 8 * half;
        size_t hbase = ((size_t)b * kN + n_glob) * (size_t)HPAD + colbase;
        size_t dbase = (((size_t)b * kS + s) * (size_t)kN + n_glob) * (size_t)CPAD;
        float v0 = acc0[v], v1 = acc1[v], v2 = acc2[v];
        int c0 = ml, c1 = 16 + ml, c2 = 32 + ml;
        h[hbase + c0] = v0;              // c0 <= 15 < 34 always
        h[hbase + c1] = v1;              // c1 <= 31 < 34 always
        if (c2 < CCAT) h[hbase + c2] = v2;
        dst[dbase + c0] = (__bf16)v0;
        dst[dbase + c1] = (__bf16)v1;
        dst[dbase + c2] = (__bf16)v2;
    }
}

// ---------------- projection + sigmoid -> r,u ----------------
__global__ __launch_bounds__(256) void proj_ru_kernel(
    const float* __restrict__ h, const float* __restrict__ W,
    const float* __restrict__ bias, float* __restrict__ ru)
{
    int idx = blockIdx.x * blockDim.x + threadIdx.x;
    if (idx >= kB * kN * 2 * kH) return;
    int o  = idx & (2 * kH - 1);
    int bn = idx >> 6;
    const float* hrow = h + (size_t)bn * HPAD;
    float acc = bias[o];
    for (int f = 0; f < FTOT; ++f)
        acc += hrow[f] * W[(size_t)f * (2 * kH) + o];
    ru[idx] = 1.0f / (1.0f + __expf(-acc));
}

// ---------------- projection + tanh + GRU gate -> outputs (x2) ----------------
__global__ __launch_bounds__(256) void proj_c_kernel(
    const float* __restrict__ h, const float* __restrict__ W,
    const float* __restrict__ bias, const float* __restrict__ ru,
    const float* __restrict__ states, float* __restrict__ out)
{
    int idx = blockIdx.x * blockDim.x + threadIdx.x;
    if (idx >= kB * kN * kH) return;
    int o  = idx & (kH - 1);
    int bn = idx >> 5;
    const float* hrow = h + (size_t)bn * HPAD;
    float acc = bias[o];
    for (int f = 0; f < FTOT; ++f)
        acc += hrow[f] * W[(size_t)f * kH + o];
    float c  = tanhf(acc);
    float u  = ru[(size_t)bn * (2 * kH) + kH + o];
    float st = states[(size_t)bn * kH + o];
    float val = u * st + (1.0f - u) * c;
    out[idx] = val;
    out[(size_t)kB * kN * kH + idx] = val;   // tuple returns (outputs, outputs)
}

// ---------------- launch ----------------
extern "C" void kernel_launch(void* const* d_in, const int* in_sizes, int n_in,
                              void* d_out, int out_size, void* d_ws, size_t ws_size,
                              hipStream_t stream) {
    const float* inputs   = (const float*)d_in[0];
    const float* supports = (const float*)d_in[1];
    const float* states   = (const float*)d_in[2];
    const float* W_ru     = (const float*)d_in[3];
    const float* b_ru     = (const float*)d_in[4];
    const float* W_c      = (const float*)d_in[5];
    const float* b_c      = (const float*)d_in[6];
    float* out = (float*)d_out;

    char* ws = (char*)d_ws;
    size_t off = 0;
    float*  h    = (float*)(ws + off);  off += (size_t)kB * kN * HPAD * sizeof(float);
    __bf16* xbf  = (__bf16*)(ws + off); off += (size_t)kB * kN * CPAD * sizeof(__bf16);
    __bf16* ping = (__bf16*)(ws + off); off += (size_t)kB * kS * kN * CPAD * sizeof(__bf16);
    __bf16* pong = (__bf16*)(ws + off); off += (size_t)kB * kS * kN * CPAD * sizeof(__bf16);
    float*  ru   = (float*)(ws + off);  off += (size_t)kB * kN * 2 * kH * sizeof(float);

    dim3 gGemm(kN / 128, kS, kB);
    int nBN = kB * kN;

    // ---- pass 1: r,u gates ----
    concat_kernel<<<(nBN + 255) / 256, 256, 0, stream>>>(inputs, states, nullptr, h, xbf);
    gemm_step_kernel<<<gGemm, 256, 0, stream>>>(supports, xbf, 0, h, ping, 0);
    gemm_step_kernel<<<gGemm, 256, 0, stream>>>(supports, ping, 1, h, pong, 1);
    proj_ru_kernel<<<(nBN * 2 * kH + 255) / 256, 256, 0, stream>>>(h, W_ru, b_ru, ru);

    // ---- pass 2: candidate c and GRU output ----
    concat_kernel<<<(nBN + 255) / 256, 256, 0, stream>>>(inputs, states, ru, h, xbf);
    gemm_step_kernel<<<gGemm, 256, 0, stream>>>(supports, xbf, 0, h, ping, 0);
    gemm_step_kernel<<<gGemm, 256, 0, stream>>>(supports, ping, 1, h, pong, 1);
    proj_c_kernel<<<(nBN * kH + 255) / 256, 256, 0, stream>>>(h, W_c, b_c, ru, states, out);
}
